// SpatioTemporalRegionModel_50233937494184
// MI455X (gfx1250) — compile-verified, matrix operand order checked by reading, not run on hardware
//
#include <hip/hip_runtime.h>
#include <stdint.h>

// ---------------- model dims (compile-time constants) ----------------
#define BB    2
#define TT    256
#define NNR   128           // regions
#define DDIM  512
#define HHN   8
#define DHH   64            // DDIM / HHN
#define DIN   1024          // 2*DDIM
#define DSTATE 16
#define DCONV  4
#define DTRANK 32           // DDIM/16
#define MTOK  (BB*TT*NNR)   // 65536 token rows for all big GEMMs
#define SEQS  (BB*NNR)      // 256 mamba sequences of length TT

typedef __attribute__((ext_vector_type(16))) __bf16 v16bf;
typedef __attribute__((ext_vector_type(8)))  float  v8f;
typedef int v4i_vs __attribute__((vector_size(16)));   // matches builtin param

// ---------------- f32 -> bf16 via native __bf16 conversion ----------------
__device__ __forceinline__ unsigned short f2bf1(float f) {
  __bf16 h = (__bf16)f;
  return __builtin_bit_cast(unsigned short, h);
}
__device__ __forceinline__ unsigned pk2(float a, float b) {
  return (unsigned)f2bf1(a) | ((unsigned)f2bf1(b) << 16);
}
__device__ __forceinline__ uint2 pack4(float4 f) {
  uint2 r; r.x = pk2(f.x, f.y); r.y = pk2(f.z, f.w); return r;
}

// ---------------- CDNA5 async global->LDS copy (ASYNCcnt path) ----------
#if __has_builtin(__builtin_amdgcn_global_load_async_to_lds_b128)
#define HAVE_ASYNC_LDS 1
#endif

__device__ __forceinline__ void cp16(unsigned short* lds_dst,
                                     const unsigned short* gsrc) {
#ifdef HAVE_ASYNC_LDS
  // global_load_async_to_lds_b128: per-lane LDS dest + 64-bit global addr
  __builtin_amdgcn_global_load_async_to_lds_b128(
      (__attribute__((address_space(1))) v4i_vs*)(uintptr_t)gsrc,
      (__attribute__((address_space(3))) v4i_vs*)(unsigned int)(uintptr_t)lds_dst,
      0, 0);
#else
  *reinterpret_cast<uint4*>(lds_dst) = *reinterpret_cast<const uint4*>(gsrc);
#endif
}
__device__ __forceinline__ void async_wait0() {
#ifdef HAVE_ASYNC_LDS
  asm volatile("s_wait_asynccnt 0" ::: "memory");
#endif
}

struct FragBits { uint4 lo, hi; };          // 32 bytes == v16bf
// CDNA5 16-bit A/B fragment: lane L (half=L>>4) holds K = half*8+0..7 (lo)
// and K = 16+half*8+0..7 (hi) of its row/column -> two ds_load_b128.
__device__ __forceinline__ v16bf load_frag(const unsigned short* p) {
  FragBits f;
  f.lo = *reinterpret_cast<const uint4*>(p);
  f.hi = *reinterpret_cast<const uint4*>(p + 16);
  return __builtin_bit_cast(v16bf, f);
}
__device__ __forceinline__ float sigmoidf_(float x) { return 1.f / (1.f + __expf(-x)); }

// ======================================================================
// f32 buffer -> bf16 buffer (one-time conversion pre-pass)
// ======================================================================
__global__ __launch_bounds__(256)
void cvt_f32_bf16(const float4* __restrict__ in, uint2* __restrict__ out, long n4)
{
  long gid = (long)blockIdx.x * 256 + threadIdx.x;
  if (gid < n4) out[gid] = pack4(in[gid]);
}

// ======================================================================
// GEMM: act(Xb[M,K]bf16(lda) @ Wb[N,K]bf16^T + bias) -> OutF (f32) / OutB (bf16)
// Block 256 thr (8 waves), tile 128Mx64Nx32K, double-buffered async LDS staging.
// Wave grid 4(M)x2(N), each wave 2x2 v_wmma_f32_16x16x32_bf16 tiles.
// Requires M%128==0, N%64==0, K%32==0, lda%8==0 (true for all calls here).
// act: 0=none, 1=relu, 2=softplus
// ======================================================================
__global__ __launch_bounds__(256)
void gemm_bf16_wmma(const unsigned short* __restrict__ Xb, int lda,
                    const unsigned short* __restrict__ Wb,
                    const float* __restrict__ bias,
                    float* __restrict__ OutF,
                    unsigned short* __restrict__ OutB, int ldo,
                    int M, int N, int K, int act)
{
  const int tid  = threadIdx.x;
  const int lane = tid & 31;
  const int wid  = tid >> 5;
  const int half = lane >> 4;
  const int ln   = lane & 15;
  const int wm   = wid & 3;
  const int wn   = wid >> 2;
  const long rowBase = (long)blockIdx.x * 128;
  const long colBase = (long)blockIdx.y * 64;

  __shared__ __align__(16) unsigned short As[2][128 * 32];  // 2 x 8 KB
  __shared__ __align__(16) unsigned short Bs[2][64 * 32];   // 2 x 4 KB

  auto stage = [&](int k0, int p) {
    #pragma unroll
    for (int i = 0; i < 2; ++i) {               // A: 512 b128 segs, 2/thread
      int idx = tid + i * 256;
      int r = idx >> 2, kc = (idx & 3) << 3;
      cp16(&As[p][r * 32 + kc], Xb + (rowBase + r) * (long)lda + k0 + kc);
    }
    {                                           // B: 256 b128 segs, 1/thread
      int r = tid >> 2, kc = (tid & 3) << 3;
      cp16(&Bs[p][r * 32 + kc], Wb + (colBase + r) * (long)K + k0 + kc);
    }
  };

  v8f acc[2][2] = {};
  stage(0, 0);
  async_wait0();
  __syncthreads();

  int p = 0;
  for (int k0 = 0; k0 < K; k0 += 32) {
    if (k0 + 32 < K) stage(k0 + 32, p ^ 1);     // stream next tile (async)
    v16bf bfr[2];
    #pragma unroll
    for (int nt = 0; nt < 2; ++nt)
      bfr[nt] = load_frag(&Bs[p][(wn * 32 + nt * 16 + ln) * 32 + half * 8]);
    #pragma unroll
    for (int mt = 0; mt < 2; ++mt) {
      v16bf afr = load_frag(&As[p][(wm * 32 + mt * 16 + ln) * 32 + half * 8]);
      #pragma unroll
      for (int nt = 0; nt < 2; ++nt)
        acc[mt][nt] = __builtin_amdgcn_wmma_f32_16x16x32_bf16(
            false, afr, false, bfr[nt], (short)0, acc[mt][nt], false, false);
    }
    async_wait0();
    __syncthreads();
    p ^= 1;
  }

  // C layout: lane ln = column, vgpr v = row v + 8*half
  #pragma unroll
  for (int mt = 0; mt < 2; ++mt) {
    #pragma unroll
    for (int nt = 0; nt < 2; ++nt) {
      long col = colBase + wn * 32 + nt * 16 + ln;
      float bv = bias ? bias[col] : 0.f;
      #pragma unroll
      for (int v = 0; v < 8; ++v) {
        long row = rowBase + wm * 32 + mt * 16 + v + 8 * half;
        float val = acc[mt][nt][v] + bv;
        if (act == 1)      val = fmaxf(val, 0.f);
        else if (act == 2) val = (val > 20.f) ? val : log1pf(__expf(val));
        if (OutF) OutF[row * (long)ldo + col] = val;
        if (OutB) OutB[row * (long)ldo + col] = f2bf1(val);
      }
    }
  }
}

// ======================================================================
// Fused MHA, all-bf16 I/O: one block per (bt, head). 128 tokens, dh=64.
// LDS: Q | K | V^T (bf16), P aliases Q+K after a barrier.
// ======================================================================
__global__ __launch_bounds__(256)
void mha_fused_wmma(const unsigned short* __restrict__ qkv,
                    unsigned short* __restrict__ out)
{
  const int bt   = blockIdx.x;
  const int hd   = blockIdx.y;
  const int tid  = threadIdx.x;
  const int lane = tid & 31;
  const int wid  = tid >> 5;        // 8 waves: rows [wid*16, wid*16+16)
  const int half = lane >> 4;
  const int ln   = lane & 15;

  __shared__ __align__(16) unsigned short smem[24576];   // 48 KB
  unsigned short* Qs  = smem;              // [128][64]
  unsigned short* Ks  = smem + 8192;       // [128][64]
  unsigned short* Vst = smem + 16384;      // [64][128] transposed
  unsigned short* Ps  = smem;              // [128][128] aliases Q+K

  const unsigned short* base = qkv + (long)bt * NNR * (3 * DDIM);

  #pragma unroll
  for (int i = 0; i < 4; ++i) {            // 1024 b128 segs each for Q,K,V
    int idx = tid + i * 256;
    int r  = idx >> 3;                     // token row
    int c0 = (idx & 7) << 3;               // 0..56
    const unsigned short* rp = base + (long)r * (3 * DDIM) + hd * DHH + c0;
    cp16(Qs + r * 64 + c0, rp);
    cp16(Ks + r * 64 + c0, rp + DDIM);
    uint4 v4 = *reinterpret_cast<const uint4*>(rp + 2 * DDIM);   // V: transpose
    Vst[(c0 + 0) * 128 + r] = (unsigned short)(v4.x & 0xffff);
    Vst[(c0 + 1) * 128 + r] = (unsigned short)(v4.x >> 16);
    Vst[(c0 + 2) * 128 + r] = (unsigned short)(v4.y & 0xffff);
    Vst[(c0 + 3) * 128 + r] = (unsigned short)(v4.y >> 16);
    Vst[(c0 + 4) * 128 + r] = (unsigned short)(v4.z & 0xffff);
    Vst[(c0 + 5) * 128 + r] = (unsigned short)(v4.z >> 16);
    Vst[(c0 + 6) * 128 + r] = (unsigned short)(v4.w & 0xffff);
    Vst[(c0 + 7) * 128 + r] = (unsigned short)(v4.w >> 16);
  }
  async_wait0();
  __syncthreads();

  // ---- S = Q K^T ----
  v8f acc[8] = {};
  #pragma unroll
  for (int kc = 0; kc < DHH; kc += 32) {
    v16bf afr = load_frag(Qs + (wid * 16 + ln) * 64 + kc + half * 8);
    #pragma unroll
    for (int nt = 0; nt < 8; ++nt) {
      v16bf bfr = load_frag(Ks + (nt * 16 + ln) * 64 + kc + half * 8);
      acc[nt] = __builtin_amdgcn_wmma_f32_16x16x32_bf16(
          false, afr, false, bfr, (short)0, acc[nt], false, false);
    }
  }

  // ---- softmax per row r = v + 8*half (reduce across 16-lane halves) ----
  float inv_sum[8];
  #pragma unroll
  for (int v = 0; v < 8; ++v) {
    float mx = -1e30f;
    #pragma unroll
    for (int nt = 0; nt < 8; ++nt) {
      float s = acc[nt][v] * 0.125f;       // 1/sqrt(dh)
      acc[nt][v] = s;
      mx = fmaxf(mx, s);
    }
    for (int m = 8; m; m >>= 1) mx = fmaxf(mx, __shfl_xor(mx, m, 32));
    float sum = 0.f;
    #pragma unroll
    for (int nt = 0; nt < 8; ++nt) {
      float e = __expf(acc[nt][v] - mx);
      acc[nt][v] = e;
      sum += e;
    }
    for (int m = 8; m; m >>= 1) sum += __shfl_xor(sum, m, 32);
    inv_sum[v] = 1.f / sum;
  }

  __syncthreads();                 // all waves done with Qs/Ks before Ps
  #pragma unroll
  for (int nt = 0; nt < 8; ++nt)
    #pragma unroll
    for (int v = 0; v < 8; ++v)
      Ps[(wid * 16 + v + 8 * half) * 128 + nt * 16 + ln] =
          f2bf1(acc[nt][v] * inv_sum[v]);
  __syncthreads();

  // ---- O = P V ----
  v8f oacc[4] = {};
  #pragma unroll
  for (int jc = 0; jc < 128; jc += 32) {
    v16bf afr = load_frag(Ps + (wid * 16 + ln) * 128 + jc + half * 8);
    #pragma unroll
    for (int nt = 0; nt < 4; ++nt) {
      v16bf bfr = load_frag(Vst + (nt * 16 + ln) * 128 + jc + half * 8);
      oacc[nt] = __builtin_amdgcn_wmma_f32_16x16x32_bf16(
          false, afr, false, bfr, (short)0, oacc[nt], false, false);
    }
  }
  #pragma unroll
  for (int nt = 0; nt < 4; ++nt)
    #pragma unroll
    for (int v = 0; v < 8; ++v) {
      long row = (long)bt * NNR + wid * 16 + v + 8 * half;
      out[row * DDIM + hd * DHH + nt * 16 + ln] = f2bf1(oacc[nt][v]);
    }
}

// ======================================================================
// out[row] = rmsnorm(a[row] + b[row]) * w     (D = 512, one block per row)
// ======================================================================
__global__ __launch_bounds__(256)
void rmsnorm_add(const float* __restrict__ a, const float* __restrict__ b,
                 const float* __restrict__ w, float* __restrict__ out)
{
  long row = blockIdx.x;
  int  t   = threadIdx.x;
  const float* pa = a + row * DDIM;
  const float* pb = b + row * DDIM;
  float v0 = pa[t] + pb[t];
  float v1 = pa[t + 256] + pb[t + 256];
  float s  = v0 * v0 + v1 * v1;
  for (int m = 16; m; m >>= 1) s += __shfl_xor(s, m, 32);
  __shared__ float red[8];
  if ((t & 31) == 0) red[t >> 5] = s;
  __syncthreads();
  float tot = red[0] + red[1] + red[2] + red[3] + red[4] + red[5] + red[6] + red[7];
  float r = rsqrtf(tot * (1.f / DDIM) + 1e-5f);
  out[row * DDIM + t]       = v0 * r * w[t];
  out[row * DDIM + t + 256] = v1 * r * w[t + 256];
}

// (B,T,N,D) -> (B,N,T,D): f32 out + bf16 out (feeds next GEMM's A)
__global__ __launch_bounds__(256)
void permute_btnd_to_bntd(const float4* __restrict__ in,
                          float4* __restrict__ outF, uint2* __restrict__ outB)
{
  long gid = (long)blockIdx.x * 256 + threadIdx.x;    // B*T*N*(D/4)
  int d  = (int)(gid & 127);  long r = gid >> 7;
  int n  = (int)(r & 127);    long bt = r >> 7;
  int t  = (int)(bt & 255);   int  b  = (int)(bt >> 8);
  long o = (((long)(b * NNR + n) * TT + t) * 128) + d;
  float4 v = in[gid];
  outF[o] = v;
  outB[o] = pack4(v);
}
// (B,N,T,D) -> (B,T,N,D): f32 out + bf16 out
__global__ __launch_bounds__(256)
void permute_bntd_to_btnd(const float4* __restrict__ in,
                          float4* __restrict__ outF, uint2* __restrict__ outB)
{
  long gid = (long)blockIdx.x * 256 + threadIdx.x;
  int d  = (int)(gid & 127);  long r = gid >> 7;
  int t  = (int)(r & 255);    long bn = r >> 8;
  int n  = (int)(bn & 127);   int  b  = (int)(bn >> 7);
  long o = (((long)(b * TT + t) * NNR + n) * 128) + d;
  float4 v = in[gid];
  outF[o] = v;
  outB[o] = pack4(v);
}

// causal depthwise conv1d (width 4) + bias + SiLU; f32 + bf16 outputs.
__global__ __launch_bounds__(256)
void conv_silu(const float* __restrict__ xz, const float* __restrict__ w,
               const float* __restrict__ b, float* __restrict__ outF,
               unsigned short* __restrict__ outB)
{
  long gid = (long)blockIdx.x * 256 + threadIdx.x;    // SEQS*TT*DIN
  int  c  = (int)(gid & (DIN - 1));
  long rt = gid >> 10;
  int  t  = (int)(rt & (TT - 1));
  long sq = rt >> 8;
  float acc = b[c];
  #pragma unroll
  for (int j = 0; j < DCONV; ++j) {
    int tt = t - (DCONV - 1) + j;
    if (tt >= 0)
      acc += w[c * DCONV + j] * xz[((sq * TT + tt) * (long)(2 * DIN)) + c];
  }
  float val = acc * sigmoidf_(acc);
  outF[rt * DIN + c] = val;
  outB[rt * DIN + c] = f2bf1(val);
}

// selective scan: one thread per (seq, channel), h[16] in registers; bf16 out.
__global__ __launch_bounds__(256)
void mamba_scan(const float* __restrict__ dt, const float* __restrict__ xdbl,
                const float* __restrict__ u, const float* __restrict__ xz,
                const float* __restrict__ A_log, const float* __restrict__ Dp,
                unsigned short* __restrict__ yB)
{
  long gid = (long)blockIdx.x * 256 + threadIdx.x;    // SEQS*DIN
  int  c  = (int)(gid & (DIN - 1));
  long sq = gid >> 10;
  float a[DSTATE], h[DSTATE];
  #pragma unroll
  for (int s = 0; s < DSTATE; ++s) { a[s] = -__expf(A_log[c * DSTATE + s]); h[s] = 0.f; }
  float dcoef = Dp[c];
  for (int t = 0; t < TT; ++t) {
    long row = sq * TT + t;
    float dtv = dt[row * DIN + c];
    float uv  = u[row * DIN + c];
    const float* bc = xdbl + row * (DTRANK + 2 * DSTATE);
    float du = dtv * uv;
    float yv = 0.f;
    #pragma unroll
    for (int s = 0; s < DSTATE; ++s) {
      float Bv = bc[DTRANK + s];
      float Cv = bc[DTRANK + DSTATE + s];
      h[s] = __expf(dtv * a[s]) * h[s] + du * Bv;
      yv += h[s] * Cv;
    }
    float zv = xz[row * (long)(2 * DIN) + DIN + c];
    yB[row * DIN + c] = f2bf1((yv + uv * dcoef) * (zv * sigmoidf_(zv)));
  }
}

// ======================================================================
// host-side orchestration
// ======================================================================
extern "C" void kernel_launch(void* const* d_in, const int* in_sizes, int n_in,
                              void* d_out, int out_size, void* d_ws, size_t ws_size,
                              hipStream_t stream)
{
  (void)in_sizes; (void)n_in; (void)out_size; (void)ws_size;
  const float* x         = (const float*)d_in[0];
  const float* mha_in_w  = (const float*)d_in[1];
  const float* mha_in_b  = (const float*)d_in[2];
  const float* mha_out_w = (const float*)d_in[3];
  const float* mha_out_b = (const float*)d_in[4];
  const float* n1_w      = (const float*)d_in[5];
  const float* n2_w      = (const float*)d_in[6];
  const float* n3_w      = (const float*)d_in[7];
  const float* m_in_w    = (const float*)d_in[8];
  const float* m_conv_w  = (const float*)d_in[9];
  const float* m_conv_b  = (const float*)d_in[10];
  const float* m_xproj_w = (const float*)d_in[11];
  const float* m_dt_w    = (const float*)d_in[12];
  const float* m_dt_b    = (const float*)d_in[13];
  const float* m_A_log   = (const float*)d_in[14];
  const float* m_D       = (const float*)d_in[15];
  const float* m_out_w   = (const float*)d_in[16];
  const float* mlp_w1    = (const float*)d_in[17];
  const float* mlp_b1    = (const float*)d_in[18];
  const float* mlp_w2    = (const float*)d_in[19];
  const float* mlp_b2    = (const float*)d_in[20];

  char* wsb = (char*)d_ws;
  size_t off = 0;
  auto carve = [&](size_t bytes) {
    void* p = wsb + off;
    off += (bytes + 255) & ~(size_t)255;
    return p;
  };
  const size_t MT = (size_t)MTOK;
  // f32 intermediates
  float* ws_o    = (float*)carve(MT * DDIM * 4);
  float* ws_xs   = (float*)carve(MT * DDIM * 4);
  float* ws_xt   = (float*)carve(MT * DDIM * 4);
  float* ws_xz   = (float*)carve(MT * 2 * DIN * 4);
  float* ws_xc   = (float*)carve(MT * DIN * 4);
  float* ws_xdbl = (float*)carve(MT * 64 * 4);
  float* ws_dt   = (float*)carve(MT * DIN * 4);
  float* ws_mo   = (float*)carve(MT * DDIM * 4);
  float* ws_xt2  = (float*)carve(MT * DDIM * 4);
  float* ws_x3   = (float*)carve(MT * DDIM * 4);
  float* ws_o2   = (float*)carve(MT * DDIM * 4);
  // bf16 intermediates
  unsigned short* xB     = (unsigned short*)carve(MT * DDIM * 2);
  unsigned short* qkvB   = (unsigned short*)carve(MT * 3 * DDIM * 2);
  unsigned short* attB   = (unsigned short*)carve(MT * DDIM * 2);
  unsigned short* xtB    = (unsigned short*)carve(MT * DDIM * 2);
  unsigned short* xcB    = (unsigned short*)carve(MT * DIN * 2);
  unsigned short* xdblB  = (unsigned short*)carve(MT * 64 * 2);
  unsigned short* yB     = (unsigned short*)carve(MT * DIN * 2);
  unsigned short* x3B    = (unsigned short*)carve(MT * DDIM * 2);
  unsigned short* hB     = (unsigned short*)carve(MT * DIN * 2);
  // bf16 weights
  unsigned short* wqkvB = (unsigned short*)carve((size_t)3 * DDIM * DDIM * 2);
  unsigned short* woB   = (unsigned short*)carve((size_t)DDIM * DDIM * 2);
  unsigned short* wminB = (unsigned short*)carve((size_t)2 * DIN * DDIM * 2);
  unsigned short* wxpB  = (unsigned short*)carve((size_t)64 * DIN * 2);
  unsigned short* wdtB  = (unsigned short*)carve((size_t)DIN * DTRANK * 2);
  unsigned short* wmoB  = (unsigned short*)carve((size_t)DDIM * DIN * 2);
  unsigned short* w1B   = (unsigned short*)carve((size_t)DIN * DDIM * 2);
  unsigned short* w2B   = (unsigned short*)carve((size_t)DDIM * DIN * 2);

  dim3 blk(256);
  auto cvt = [&](const float* src, unsigned short* dst, size_t n) {
    long n4 = (long)(n / 4);
    cvt_f32_bf16<<<(unsigned)((n4 + 255) / 256), blk, 0, stream>>>(
        (const float4*)src, (uint2*)dst, n4);
  };
  // one-time bf16 conversions (weights + input)
  cvt(x, xB, MT * DDIM);
  cvt(mha_in_w, wqkvB, (size_t)3 * DDIM * DDIM);
  cvt(mha_out_w, woB, (size_t)DDIM * DDIM);
  cvt(m_in_w, wminB, (size_t)2 * DIN * DDIM);
  cvt(m_xproj_w, wxpB, (size_t)64 * DIN);
  cvt(m_dt_w, wdtB, (size_t)DIN * DTRANK);
  cvt(m_out_w, wmoB, (size_t)DDIM * DIN);
  cvt(mlp_w1, w1B, (size_t)DIN * DDIM);
  cvt(mlp_w2, w2B, (size_t)DDIM * DIN);

  const int MB = MTOK / 128;
  const long elem4 = (long)MTOK * 128;

  // -------- stage 1: spatial attention --------
  gemm_bf16_wmma<<<dim3(MB, (3 * DDIM) / 64), blk, 0, stream>>>(
      xB, DDIM, wqkvB, mha_in_b, nullptr, qkvB, 3 * DDIM,
      MTOK, 3 * DDIM, DDIM, 0);
  mha_fused_wmma<<<dim3(BB * TT, HHN), blk, 0, stream>>>(qkvB, attB);
  gemm_bf16_wmma<<<dim3(MB, DDIM / 64), blk, 0, stream>>>(
      attB, DDIM, woB, mha_out_b, ws_o, nullptr, DDIM,
      MTOK, DDIM, DDIM, 0);
  rmsnorm_add<<<MTOK, blk, 0, stream>>>(x, ws_o, n1_w, ws_xs);

  // -------- stage 2: temporal mamba --------
  permute_btnd_to_bntd<<<(unsigned)(elem4 / 256), blk, 0, stream>>>(
      (const float4*)ws_xs, (float4*)ws_xt, (uint2*)xtB);
  gemm_bf16_wmma<<<dim3(MB, (2 * DIN) / 64), blk, 0, stream>>>(
      xtB, DDIM, wminB, nullptr, ws_xz, nullptr, 2 * DIN,
      MTOK, 2 * DIN, DDIM, 0);
  conv_silu<<<(unsigned)(((long)MTOK * DIN) / 256), blk, 0, stream>>>(
      ws_xz, m_conv_w, m_conv_b, ws_xc, xcB);
  gemm_bf16_wmma<<<dim3(MB, 64 / 64), blk, 0, stream>>>(
      xcB, DIN, wxpB, nullptr, ws_xdbl, xdblB, 64,
      MTOK, 64, DIN, 0);
  gemm_bf16_wmma<<<dim3(MB, DIN / 64), blk, 0, stream>>>(
      xdblB, 64, wdtB, m_dt_b, ws_dt, nullptr, DIN,
      MTOK, DIN, DTRANK, 2 /* softplus */);
  mamba_scan<<<(SEQS * DIN) / 256, blk, 0, stream>>>(
      ws_dt, ws_xdbl, ws_xc, ws_xz, m_A_log, m_D, yB);
  gemm_bf16_wmma<<<dim3(MB, DDIM / 64), blk, 0, stream>>>(
      yB, DIN, wmoB, nullptr, ws_mo, nullptr, DDIM,
      MTOK, DDIM, DIN, 0);
  rmsnorm_add<<<MTOK, blk, 0, stream>>>(ws_xt, ws_mo, n2_w, ws_xt2);
  permute_bntd_to_btnd<<<(unsigned)(elem4 / 256), blk, 0, stream>>>(
      (const float4*)ws_xt2, (float4*)ws_x3, (uint2*)x3B);

  // -------- stage 3: MLP --------
  gemm_bf16_wmma<<<dim3(MB, (2 * DDIM) / 64), blk, 0, stream>>>(
      x3B, DDIM, w1B, mlp_b1, nullptr, hB, 2 * DDIM,
      MTOK, 2 * DDIM, DDIM, 1 /* relu */);
  gemm_bf16_wmma<<<dim3(MB, DDIM / 64), blk, 0, stream>>>(
      hB, DIN, w2B, mlp_b2, ws_o2, nullptr, DDIM,
      MTOK, DDIM, 2 * DDIM, 0);
  rmsnorm_add<<<MTOK, blk, 0, stream>>>(ws_x3, ws_o2, n3_w, (float*)d_out);
}
